// DLRM_Net_53377853555315
// MI455X (gfx1250) — compile-verified, matrix operand order checked by reading, not run on hardware
//
#include <hip/hip_runtime.h>
#include <stdint.h>

typedef __attribute__((ext_vector_type(16))) __bf16 v16bf;
typedef __attribute__((ext_vector_type(8)))  float  v8f;

#define BATCH  32768
#define NTABS  26
#define EMBDIM 128
#define EROWS  50000

union FragBF { uint32_t u[8]; v16bf v; };

// round-to-nearest-even fp32 -> bf16 (bit trick; avoids relying on __bf16 arithmetic)
__device__ __forceinline__ uint16_t f2bf(float f) {
    uint32_t u = __float_as_uint(f);
    uint32_t r = u + 0x7FFFu + ((u >> 16) & 1u);
    return (uint16_t)(r >> 16);
}

// Load a 16x32 bf16 fragment (A-layout per CDNA5 ISA 7.12.2) from an LDS tile.
// Lane L holds row M = rowBase + (L%16); uint pair index per VGPR v:
//   (v&3) + ((v>>2)<<3) + ((L>>4)<<2)
__device__ __forceinline__ FragBF lds_frag(const uint32_t* base, int rowBase, int lane, int rowStrideU32) {
    FragBF f;
    const uint32_t* p = base + (size_t)(rowBase + (lane & 15)) * rowStrideU32;
    const int halfOff = (lane >> 4) << 2;
#pragma unroll
    for (int v = 0; v < 8; ++v) {
        f.u[v] = p[(v & 3) + ((v >> 2) << 3) + halfOff];
    }
    return f;
}

// Issue two async 16-byte global->LDS copies (32 B total) for this lane.
__device__ __forceinline__ void async_copy32(uint32_t ldsOff, const void* gp) {
    asm volatile("global_load_async_to_lds_b128 %0, %1, off\n\t"
                 "global_load_async_to_lds_b128 %0, %1, off offset:16"
                 :: "v"(ldsOff), "v"(gp) : "memory");
}

__device__ __forceinline__ void wait_async0() {
    asm volatile("s_wait_asynccnt 0x0" ::: "memory");
}

// ---------------------------------------------------------------------------
// Generic tiled GEMM:  out[B x N] = act( A[B x Kp](bf16) * W[N x Kp]^T + bias )
// 128x128 tile per workgroup, 8 wave32 waves (2x4), each wave 64x32 via
// 4x2 v_wmma_f32_16x16x32_bf16 accumulators.
// Double-buffered LDS; tiles staged with GLOBAL_LOAD_ASYNC_TO_LDS_B128 so the
// global->LDS copy of tile k+1 overlaps the WMMAs of tile k (ASYNCcnt fence).
// ---------------------------------------------------------------------------
__global__ __launch_bounds__(256)
void gemm_bf16(const uint16_t* __restrict__ A,
               const uint16_t* __restrict__ W,
               const float* __restrict__ bias,
               uint16_t* __restrict__ obf,
               float* __restrict__ of32,
               int Kp, int N, int ldo, int act)   // act: 0=relu->bf16, 1=sigmoid->f32
{
    __shared__ uint32_t sA[2][128 * 16];   // 2 x (128 rows x 32 bf16)
    __shared__ uint32_t sW[2][128 * 16];

    const int tid  = threadIdx.x;
    const int lane = tid & 31;
    const int wave = tid >> 5;
    const int wm   = wave & 1;          // 2 row-groups of 64
    const int wn   = wave >> 1;         // 4 col-groups of 32
    const int m0   = blockIdx.x * 128;
    const int n0   = blockIdx.y * 128;

    const int ldRow  = tid >> 1;        // 2 threads per tile row
    const int ldHalf = tid & 1;         // 16 bf16 (32 B) each

    const bool fullN = (n0 + 128) <= N; // async path legal (no zero-fill needed)

    const uint32_t stOff = (uint32_t)(ldRow * 64 + ldHalf * 32);
    const uint32_t ldsAoff[2] = { (uint32_t)(uintptr_t)sA[0] + stOff,
                                  (uint32_t)(uintptr_t)sA[1] + stOff };
    const uint32_t ldsWoff[2] = { (uint32_t)(uintptr_t)sW[0] + stOff,
                                  (uint32_t)(uintptr_t)sW[1] + stOff };

    auto stage = [&](int buf, int k0) {
        const uint16_t* gA = A + (size_t)(m0 + ldRow) * Kp + k0 + ldHalf * 16;
        async_copy32(ldsAoff[buf], gA);                   // A tile: always full
        if (fullN) {
            const uint16_t* gW = W + (size_t)(n0 + ldRow) * Kp + k0 + ldHalf * 16;
            async_copy32(ldsWoff[buf], gW);
        } else {
            uint4 w0 = make_uint4(0u, 0u, 0u, 0u), w1 = w0;
            if (n0 + ldRow < N) {
                const uint4* gw = (const uint4*)(W + (size_t)(n0 + ldRow) * Kp + k0 + ldHalf * 16);
                w0 = gw[0]; w1 = gw[1];
            }
            uint4* dw = ((uint4*)sW[buf]) + ldRow * 4 + ldHalf * 2;
            dw[0] = w0; dw[1] = w1;
        }
    };

    v8f acc[4][2] = {};

    const int nk = Kp >> 5;             // k-steps of 32
    stage(0, 0);                        // prologue: tile 0 in flight

    for (int ki = 0; ki < nk; ++ki) {
        const int cur = ki & 1;
        wait_async0();                  // tile ki landed in LDS (async in-order)
        __syncthreads();                // all waves past reads of the other buffer

        if (ki + 1 < nk) stage(cur ^ 1, (ki + 1) << 5);   // overlaps WMMAs below

        FragBF bf0 = lds_frag(sW[cur], wn * 32 +  0, lane, 16);
        FragBF bf1 = lds_frag(sW[cur], wn * 32 + 16, lane, 16);
#pragma unroll
        for (int mt = 0; mt < 4; ++mt) {
            FragBF af = lds_frag(sA[cur], wm * 64 + mt * 16, lane, 16);
            acc[mt][0] = __builtin_amdgcn_wmma_f32_16x16x32_bf16(false, af.v, false, bf0.v,
                                                                 (short)0, acc[mt][0], false, false);
            acc[mt][1] = __builtin_amdgcn_wmma_f32_16x16x32_bf16(false, af.v, false, bf1.v,
                                                                 (short)0, acc[mt][1], false, false);
        }
    }

    // C/D layout: VGPR r -> M = r + 8*(lane/16), N = lane%16
    const int half  = lane >> 4;
    const int col16 = lane & 15;
#pragma unroll
    for (int nt = 0; nt < 2; ++nt) {
        const int col = n0 + wn * 32 + nt * 16 + col16;
        if (col >= N) continue;
        const float bv = bias[col];
#pragma unroll
        for (int mt = 0; mt < 4; ++mt) {
#pragma unroll
            for (int r = 0; r < 8; ++r) {
                const int row = m0 + wm * 64 + mt * 16 + half * 8 + r;
                float x = acc[mt][nt][r] + bv;
                x = (act == 0) ? fmaxf(x, 0.0f) : (1.0f / (1.0f + __expf(-x)));
                if (act == 0) obf[(size_t)row * ldo + col] = f2bf(x);
                else          of32[(size_t)row * ldo + col] = x;
            }
        }
    }
}

// ---------------------------------------------------------------------------
// Fused embedding-gather + interaction (one wave per sample).
// ---------------------------------------------------------------------------
__global__ __launch_bounds__(128)
void interact_wmma(const uint16_t* __restrict__ x3,
                   const int* __restrict__ lsi,
                   const float* __restrict__ emb,
                   uint16_t* __restrict__ R)
{
    __shared__ uint32_t sT[4 * 32 * 64];  // 4 waves x (32 rows x 128 bf16) = 32 KB
    const int wave = threadIdx.x >> 5;
    const int lane = threadIdx.x & 31;
    const int b    = blockIdx.x * 4 + wave;

    uint32_t* T  = sT + wave * (32 * 64);
    uint2*    Tv = (uint2*)T;             // 32 uint2 per row (4 bf16 each)

    const uint16_t* xrow = x3 + (size_t)b * EMBDIM;
    Tv[lane] = ((const uint2*)xrow)[lane];                   // row 0 = bottom-MLP output

    int idx = lsi[b];                                        // table 0
    for (int t = 0; t < NTABS; ++t) {
        const float4* e = (const float4*)(emb + ((size_t)t * EROWS + (size_t)idx) * EMBDIM);
        if (t + 1 < NTABS) {
            idx = lsi[(size_t)(t + 1) * BATCH + b];          // next table's row
            __builtin_prefetch(emb + ((size_t)(t + 1) * EROWS + (size_t)idx) * EMBDIM, 0, 0);
        }
        float4 f = e[lane];
        uint32_t lo = (uint32_t)f2bf(f.x) | ((uint32_t)f2bf(f.y) << 16);
        uint32_t hi = (uint32_t)f2bf(f.z) | ((uint32_t)f2bf(f.w) << 16);
        Tv[(t + 1) * 32 + lane] = make_uint2(lo, hi);
    }
#pragma unroll
    for (int rr = 27; rr < 32; ++rr) Tv[rr * 32 + lane] = make_uint2(0u, 0u);

    asm volatile("s_wait_dscnt 0x0" ::: "memory");           // LDS RAW fence

    v8f acc[2][2] = {};
#pragma unroll
    for (int ks = 0; ks < 4; ++ks) {                         // K = 128 in 4 chunks of 32
        FragBF f0 = lds_frag(T + ks * 16,  0, lane, 64);
        FragBF f1 = lds_frag(T + ks * 16, 16, lane, 64);
        acc[0][0] = __builtin_amdgcn_wmma_f32_16x16x32_bf16(false, f0.v, false, f0.v, (short)0, acc[0][0], false, false);
        acc[0][1] = __builtin_amdgcn_wmma_f32_16x16x32_bf16(false, f0.v, false, f1.v, (short)0, acc[0][1], false, false);
        acc[1][0] = __builtin_amdgcn_wmma_f32_16x16x32_bf16(false, f1.v, false, f0.v, (short)0, acc[1][0], false, false);
        acc[1][1] = __builtin_amdgcn_wmma_f32_16x16x32_bf16(false, f1.v, false, f1.v, (short)0, acc[1][1], false, false);
    }

    uint16_t* r = R + (size_t)b * 512;
    ((uint2*)r)[lane] = ((const uint2*)xrow)[lane];          // R[:,0:128] = x3

    const int half = lane >> 4;
    const int j16  = lane & 15;
#pragma unroll
    for (int mt = 0; mt < 2; ++mt)
#pragma unroll
        for (int nt = 0; nt < 2; ++nt)
#pragma unroll
            for (int reg = 0; reg < 8; ++reg) {
                const int i = mt * 16 + half * 8 + reg;
                const int j = nt * 16 + j16;
                if (i < 27 && j < i)
                    r[128 + (i * (i - 1)) / 2 + j] = f2bf(acc[mt][nt][reg]);
            }

    for (int c = 479 + lane; c < 512; c += 32) r[c] = (uint16_t)0;   // zero K-pad for top layer 0
}

// fp32 (N x K) weights -> zero-K-padded bf16 (N x Kp)
__global__ void cvt_weight(const float* __restrict__ w, uint16_t* __restrict__ o, int N, int K, int Kp)
{
    int i = blockIdx.x * 256 + threadIdx.x;
    if (i >= N * Kp) return;
    int n = i / Kp, k = i % Kp;
    o[i] = (k < K) ? f2bf(w[(size_t)n * K + k]) : (uint16_t)0;
}

// dense_x (B x 13 fp32) -> (B x 32 bf16), zero padded
__global__ void pad_dense(const float* __restrict__ x, uint16_t* __restrict__ o)
{
    int i = blockIdx.x * 256 + threadIdx.x;    // B*32 threads
    int row = i >> 5, c = i & 31;
    o[i] = (c < 13) ? f2bf(x[(size_t)row * 13 + c]) : (uint16_t)0;
}

extern "C" void kernel_launch(void* const* d_in, const int* in_sizes, int n_in,
                              void* d_out, int out_size, void* d_ws, size_t ws_size,
                              hipStream_t stream)
{
    // setup_inputs() order: dense_x, lS_i, lS_o, emb_w, bot_w0,b0,w1,b1,w2,b2, top_w0,b0..w4,b4
    const float* dense = (const float*)d_in[0];
    const int*   lSi   = (const int*)d_in[1];
    const float* emb   = (const float*)d_in[3];

    uint8_t* ws = (uint8_t*)d_ws;
    size_t off = 0;
    auto carve = [&](size_t bytes) -> void* {
        off = (off + 255) & ~(size_t)255;
        void* p = ws + off;
        off += bytes;
        return p;
    };

    uint16_t* bufA = (uint16_t*)carve((size_t)BATCH * 1024 * 2);
    uint16_t* bufB = (uint16_t*)carve((size_t)BATCH * 1024 * 2);
    uint16_t* bufX = (uint16_t*)carve((size_t)BATCH * 128 * 2);

    struct Layer { const float* w; const float* b; int N, K, Kp; uint16_t* wb; };
    Layer ls[8] = {
        { (const float*)d_in[4],  (const float*)d_in[5],   512,   13,   32, nullptr },
        { (const float*)d_in[6],  (const float*)d_in[7],   256,  512,  512, nullptr },
        { (const float*)d_in[8],  (const float*)d_in[9],   128,  256,  256, nullptr },
        { (const float*)d_in[10], (const float*)d_in[11], 1024,  479,  512, nullptr },
        { (const float*)d_in[12], (const float*)d_in[13], 1024, 1024, 1024, nullptr },
        { (const float*)d_in[14], (const float*)d_in[15],  512, 1024, 1024, nullptr },
        { (const float*)d_in[16], (const float*)d_in[17],  256,  512,  512, nullptr },
        { (const float*)d_in[18], (const float*)d_in[19],    1,  256,  256, nullptr },
    };
    for (int i = 0; i < 8; ++i) {
        ls[i].wb = (uint16_t*)carve((size_t)ls[i].N * ls[i].Kp * 2);
        const int tot = ls[i].N * ls[i].Kp;
        cvt_weight<<<(tot + 255) / 256, 256, 0, stream>>>(ls[i].w, ls[i].wb, ls[i].N, ls[i].K, ls[i].Kp);
    }

    pad_dense<<<(BATCH * 32) / 256, 256, 0, stream>>>(dense, bufA);

    auto gemm = [&](const uint16_t* Ain, const Layer& l, uint16_t* obf, float* of32, int ldo, int act) {
        dim3 grid(BATCH / 128, (unsigned)((l.N + 127) / 128));
        gemm_bf16<<<grid, 256, 0, stream>>>(Ain, l.wb, l.b, obf, of32, l.Kp, l.N, ldo, act);
    };

    // Bottom MLP: 13->512->256->128
    gemm(bufA, ls[0], bufB, nullptr, 512, 0);
    gemm(bufB, ls[1], bufA, nullptr, 256, 0);
    gemm(bufA, ls[2], bufX, nullptr, 128, 0);

    // Fused gather + pairwise interaction -> R (B x 512, K-padded) in bufB
    interact_wmma<<<BATCH / 4, 128, 0, stream>>>(bufX, lSi, emb, bufB);

    // Top MLP: 479->1024->1024->512->256->1 (sigmoid on last)
    gemm(bufB, ls[3], bufA, nullptr, 1024, 0);
    gemm(bufA, ls[4], bufB, nullptr, 1024, 0);
    gemm(bufB, ls[5], bufA, nullptr,  512, 0);
    gemm(bufA, ls[6], bufB, nullptr,  256, 0);
    gemm(bufB, ls[7], nullptr, (float*)d_out, 1, 1);

    (void)in_sizes; (void)n_in; (void)out_size; (void)ws_size;
}
